// Attention_69973607186925
// MI455X (gfx1250) — matrix operand order repaired and checked by reading, not compiled, not run on hardware
//
#include <hip/hip_runtime.h>
#include <stdint.h>

// ---------------- common WMMA helpers (gfx1250, wave32) ----------------
typedef __attribute__((ext_vector_type(16))) __bf16 v16bf;
typedef __attribute__((ext_vector_type(8)))  float  v8f;

union FragAB { v16bf v; uint4 q[2]; };
union FragC  { v8f   v; float f[8]; };

#define WMMA_BF16(C, A, B) \
  (C).v = __builtin_amdgcn_wmma_f32_16x16x32_bf16(false, (A).v, false, (B).v, (short)0, (C).v, false, false)

#if defined(__gfx1250__) && __has_builtin(__builtin_amdgcn_global_load_async_to_lds_b128) && \
    __has_builtin(__builtin_amdgcn_s_wait_asynccnt)
#define USE_ASYNC 1
#else
#define USE_ASYNC 0
#endif

__device__ __forceinline__ unsigned short f32_bf16(float x) {
  unsigned int u = __float_as_uint(x);
  unsigned int r = (u + 0x7FFFu + ((u >> 16) & 1u)) >> 16;  // RNE
  return (unsigned short)r;
}
__device__ __forceinline__ unsigned int pack2(float a, float b) {
  return (unsigned int)f32_bf16(a) | ((unsigned int)f32_bf16(b) << 16);
}

#if USE_ASYNC
// Builtin signature (from clang diagnostic): param is
//   '__attribute__((__vector_size__(4 * sizeof(int)))) int __device__ *'
// i.e. int4 (gcc vector_size) pointers with AS1 (global) / AS3 (LDS) qualifiers.
typedef int v4i_vs __attribute__((vector_size(16)));
typedef __attribute__((address_space(1))) v4i_vs glob_v4i;
typedef __attribute__((address_space(3))) v4i_vs lds_v4i;

// async 16-byte global -> LDS copy (ASYNCcnt-tracked DMA)
__device__ __forceinline__ void async_cp16(const unsigned short* g, unsigned short* l) {
  __builtin_amdgcn_global_load_async_to_lds_b128(
      (glob_v4i*)(unsigned long long)(size_t)g,
      (lds_v4i*)(unsigned int)(size_t)l,
      0, 0);
}
#endif

// ---------------- f32 -> bf16 conversion pre-pass ----------------
__global__ __launch_bounds__(256) void cvt_bf16(const float* __restrict__ src,
                                                unsigned short* __restrict__ dst, int n4) {
  int i = blockIdx.x * 256 + threadIdx.x;
  if (i < n4) {
    float4 v = ((const float4*)src)[i];
    uint2 o; o.x = pack2(v.x, v.y); o.y = pack2(v.z, v.w);
    ((uint2*)dst)[i] = o;
  }
}

// ---------------- shared double-buffered bf16 GEMM mainloop ----------------
// A: 128 rows x 1024 (u16, row stride 1024), B^T: 128 rows x 1024. K = 1024, 32 steps.
__device__ __forceinline__ void stage_tile(const unsigned short* gA, const unsigned short* gB,
                                           unsigned short* dA, unsigned short* dB,
                                           int tid, int k0) {
#pragma unroll
  for (int it = 0; it < 2; ++it) {
    int idx = tid + it * 256;      // uint4 index 0..511 (128 rows x 4 chunks)
    int row = idx >> 2;
    int kc  = (idx & 3) * 8;
    const unsigned short* srcA = gA + (size_t)row * 1024 + k0 + kc;
    const unsigned short* srcB = gB + (size_t)row * 1024 + k0 + kc;
#if USE_ASYNC
    async_cp16(srcA, dA + row * 32 + kc);
    async_cp16(srcB, dB + row * 32 + kc);
#else
    *(uint4*)(dA + row * 32 + kc) = *(const uint4*)srcA;
    *(uint4*)(dB + row * 32 + kc) = *(const uint4*)srcB;
#endif
  }
}

__device__ __forceinline__ void gemm_mainloop(const unsigned short* gA, const unsigned short* gB,
                                              unsigned short (*sA)[128 * 32],
                                              unsigned short (*sB)[128 * 32],
                                              FragC c[2][4], int tid) {
  const int lane = tid & 31;
  const int wave = tid >> 5;
  const int wm = (wave & 3) * 32;
  const int wn = (wave >> 2) * 64;
  const int ll = lane & 15, lh = lane >> 4;
  const int half8 = lh * 8, half16 = lh * 16;

  stage_tile(gA, gB, sA[0], sB[0], tid, 0);
#pragma unroll 1
  for (int s = 0; s < 32; ++s) {
    int cur = s & 1;
    if (s + 1 < 32) stage_tile(gA, gB, sA[cur ^ 1], sB[cur ^ 1], tid, (s + 1) * 32);
#if USE_ASYNC
    if (s + 1 < 32) __builtin_amdgcn_s_wait_asynccnt(4);  // current stage's 4 copies done
    else            __builtin_amdgcn_s_wait_asynccnt(0);
#endif
    __syncthreads();

    FragAB a[2], b[4];
#pragma unroll
    for (int i = 0; i < 2; i++) {
      int row = wm + i * 16 + ll;
      a[i].q[0] = *(const uint4*)(sA[cur] + row * 32 + half8);
      a[i].q[1] = *(const uint4*)(sA[cur] + row * 32 + 16 + half8);
    }
#pragma unroll
    for (int j = 0; j < 4; j++) {
      int row = wn + j * 16 + ll;
      b[j].q[0] = *(const uint4*)(sB[cur] + row * 32 + half8);
      b[j].q[1] = *(const uint4*)(sB[cur] + row * 32 + 16 + half8);
    }
#pragma unroll
    for (int i = 0; i < 2; i++)
#pragma unroll
      for (int j = 0; j < 4; j++) WMMA_BF16(c[i][j], a[i], b[j]);
    __syncthreads();   // protect buf[cur] before next iteration overwrites it
  }
}

// ---------------- Kernel 1: qkv = xb @ Wqkv^T + bqkv, scatter to Q/K/V^T ----------------
__global__ __launch_bounds__(256) void qkv_gemm(
    const unsigned short* __restrict__ xb, const unsigned short* __restrict__ wqb,
    const float* __restrict__ bqkv,
    unsigned short* __restrict__ qw,   // (g, n, d) bf16
    unsigned short* __restrict__ kw,   // (g, n, d) bf16
    unsigned short* __restrict__ vtw)  // (g, d, n) bf16
{
  __shared__ unsigned short sA[2][128 * 32];
  __shared__ unsigned short sB[2][128 * 32];
  const int tid = threadIdx.x;
  const int mBase = blockIdx.y * 128;
  const int jBase = blockIdx.x * 128;

  FragC c[2][4];
#pragma unroll
  for (int i = 0; i < 2; i++)
#pragma unroll
    for (int j = 0; j < 4; j++)
#pragma unroll
      for (int r = 0; r < 8; r++) c[i][j].f[r] = 0.f;

  gemm_mainloop(xb + (size_t)mBase * 1024, wqb + (size_t)jBase * 1024, sA, sB, c, tid);

  const int lane = tid & 31, wave = tid >> 5;
  const int wm = (wave & 3) * 32, wn = (wave >> 2) * 64;
  const int ll = lane & 15, half8 = (lane >> 4) * 8;
#pragma unroll
  for (int i = 0; i < 2; i++) {
    int mrow0 = mBase + wm + i * 16 + half8;       // 8-aligned
    int nidx  = mrow0 >> 6;                        // constant over r
#pragma unroll
    for (int j = 0; j < 4; j++) {
      int jc0  = jBase + wn + j * 16;
      int part = __builtin_amdgcn_readfirstlane(jc0) >> 10;   // scalar: 0:Q 1:K 2:V
      int jc   = jc0 + ll;
      float bias = bqkv[jc];
      int c0 = jc & 1023;
      int h  = c0 >> 6;
      int d  = c0 & 63;
#pragma unroll
      for (int r = 0; r < 8; r++) {
        int bb = (mrow0 + r) & 63;
        int g  = bb * 16 + h;
        unsigned short bf = f32_bf16(c[i][j].f[r] + bias);
        if (part == 0)      qw [(size_t)(g * 288 + nidx) * 64 + d] = bf;
        else if (part == 1) kw [(size_t)(g * 288 + nidx) * 64 + d] = bf;
        else                vtw[(size_t)(g * 64 + d) * 288 + nidx] = bf;
      }
    }
  }
}

// ---------------- Kernel 2: per-head attention, one wave per (g, 16-query tile) ----------------
__global__ __launch_bounds__(32) void attn_kernel(
    const unsigned short* __restrict__ qw, const unsigned short* __restrict__ kw,
    const unsigned short* __restrict__ vtw,
    unsigned short* __restrict__ yw)   // (n, b, c) bf16
{
  __shared__ unsigned short sP[16 * 288];
  const int lane = threadIdx.x & 31;
  const int g  = blockIdx.y;
  const int qt = blockIdx.x;
  const int ll = lane & 15, lh = lane >> 4;
  const int half8 = lh * 8, half16 = lh * 16;

  FragAB aq[2];
  {
    const unsigned short* qrow = qw + (size_t)(g * 288 + qt * 16 + ll) * 64;
#pragma unroll
    for (int s = 0; s < 2; s++) {
      aq[s].q[0] = *(const uint4*)(qrow + s * 32 + half8);
      aq[s].q[1] = *(const uint4*)(qrow + s * 32 + 16 + half8);
    }
  }

  FragC sfrag[18];
#pragma unroll
  for (int t = 0; t < 18; t++) {
    const unsigned short* krow = kw + (size_t)(g * 288 + t * 16 + ll) * 64;
    FragAB b0, b1;
    b0.q[0] = *(const uint4*)(krow + half8);
    b0.q[1] = *(const uint4*)(krow + 16 + half8);
    b1.q[0] = *(const uint4*)(krow + 32 + half8);
    b1.q[1] = *(const uint4*)(krow + 32 + 16 + half8);
#pragma unroll
    for (int r = 0; r < 8; r++) sfrag[t].f[r] = 0.f;
    WMMA_BF16(sfrag[t], aq[0], b0);
    WMMA_BF16(sfrag[t], aq[1], b1);
  }

  float rmax[8];
#pragma unroll
  for (int r = 0; r < 8; r++) rmax[r] = -1e30f;
#pragma unroll
  for (int t = 0; t < 18; t++) {
    int nk  = t * 16 + ll;
    int nkb = nk & ~3;
#pragma unroll
    for (int r = 0; r < 8; r++) {
      int nq = qt * 16 + r + half8;
      float v = sfrag[t].f[r] * 0.125f;
      bool blocked = (nk < 32) && !((nq < 32) && (nq >= nkb));
      v = blocked ? -1e30f : v;
      sfrag[t].f[r] = v;
      rmax[r] = fmaxf(rmax[r], v);
    }
  }
#pragma unroll
  for (int r = 0; r < 8; r++)
#pragma unroll
    for (int msk = 8; msk >= 1; msk >>= 1)
      rmax[r] = fmaxf(rmax[r], __shfl_xor(rmax[r], msk, 32));

  float rsum[8];
#pragma unroll
  for (int r = 0; r < 8; r++) rsum[r] = 0.f;
#pragma unroll
  for (int t = 0; t < 18; t++)
#pragma unroll
    for (int r = 0; r < 8; r++) {
      float p = __expf(sfrag[t].f[r] - rmax[r]);
      sfrag[t].f[r] = p;
      rsum[r] += p;
    }
#pragma unroll
  for (int r = 0; r < 8; r++)
#pragma unroll
    for (int msk = 8; msk >= 1; msk >>= 1)
      rsum[r] += __shfl_xor(rsum[r], msk, 32);
  float rinv[8];
#pragma unroll
  for (int r = 0; r < 8; r++) rinv[r] = 1.f / rsum[r];

#pragma unroll
  for (int t = 0; t < 18; t++) {
    int nk = t * 16 + ll;
#pragma unroll
    for (int r = 0; r < 8; r++) {
      int m = r + half8;
      sP[m * 288 + nk] = f32_bf16(sfrag[t].f[r] * rinv[r]);
    }
  }
  __syncthreads();

  FragC o[4];
#pragma unroll
  for (int j = 0; j < 4; j++)
#pragma unroll
    for (int r = 0; r < 8; r++) o[j].f[r] = 0.f;
#pragma unroll
  for (int t = 0; t < 9; t++) {
    FragAB ap;
    const unsigned short* prow = sP + ll * 288 + t * 32;
    ap.q[0] = *(const uint4*)(prow + half8);
    ap.q[1] = *(const uint4*)(prow + 16 + half8);
#pragma unroll
    for (int j = 0; j < 4; j++) {
      FragAB bv;
      const unsigned short* vrow = vtw + (size_t)(g * 64 + j * 16 + ll) * 288 + t * 32;
      bv.q[0] = *(const uint4*)(vrow + half8);
      bv.q[1] = *(const uint4*)(vrow + 16 + half8);
      WMMA_BF16(o[j], ap, bv);
    }
  }

  int b = g >> 4, h = g & 15;
#pragma unroll
  for (int j = 0; j < 4; j++) {
    int d = j * 16 + ll;
#pragma unroll
    for (int r = 0; r < 8; r++) {
      int nq = qt * 16 + r + half8;
      yw[(size_t)(nq * 64 + b) * 1024 + h * 64 + d] = f32_bf16(o[j].f[r]);
    }
  }
}

// ---------------- Kernel 3: out = Y @ Wproj^T + bproj (f32 output) ----------------
__global__ __launch_bounds__(256) void proj_gemm(
    const unsigned short* __restrict__ yw, const unsigned short* __restrict__ wpb,
    const float* __restrict__ bproj, float* __restrict__ out)
{
  __shared__ unsigned short sA[2][128 * 32];
  __shared__ unsigned short sB[2][128 * 32];
  const int tid = threadIdx.x;
  const int mBase = blockIdx.y * 128;
  const int jBase = blockIdx.x * 128;

  FragC c[2][4];
#pragma unroll
  for (int i = 0; i < 2; i++)
#pragma unroll
    for (int j = 0; j < 4; j++)
#pragma unroll
      for (int r = 0; r < 8; r++) c[i][j].f[r] = 0.f;

  gemm_mainloop(yw + (size_t)mBase * 1024, wpb + (size_t)jBase * 1024, sA, sB, c, tid);

  const int lane = tid & 31, wave = tid >> 5;
  const int wm = (wave & 3) * 32, wn = (wave >> 2) * 64;
  const int ll = lane & 15, half8 = (lane >> 4) * 8;
#pragma unroll
  for (int i = 0; i < 2; i++) {
    int mrow0 = mBase + wm + i * 16 + half8;
#pragma unroll
    for (int j = 0; j < 4; j++) {
      int jc = jBase + wn + j * 16 + ll;
      float bias = bproj[jc];
#pragma unroll
      for (int r = 0; r < 8; r++)
        out[(size_t)(mrow0 + r) * 1024 + jc] = c[i][j].f[r] + bias;
    }
  }
}

// ---------------- launcher ----------------
extern "C" void kernel_launch(void* const* d_in, const int* in_sizes, int n_in,
                              void* d_out, int out_size, void* d_ws, size_t ws_size,
                              hipStream_t stream) {
  const float* x     = (const float*)d_in[0];   // (288, 64, 1024)
  const float* Wqkv  = (const float*)d_in[1];   // (3072, 1024)
  const float* bqkv  = (const float*)d_in[2];   // (3072,)
  const float* Wproj = (const float*)d_in[3];   // (1024, 1024)
  const float* bproj = (const float*)d_in[4];   // (1024,)
  float* out = (float*)d_out;                   // (288, 64, 1024)

  unsigned short* ws = (unsigned short*)d_ws;
  const size_t S = (size_t)1024 * 288 * 64;     // 18,874,368 elements
  unsigned short* qw  = ws;                     // Q   (g, n, d)
  unsigned short* kw  = qw + S;                 // K   (g, n, d)
  unsigned short* vtw = kw + S;                 // V^T (g, d, n)
  unsigned short* yw  = vtw + S;                // attn out (n, b, c)
  unsigned short* xb  = yw + S;                 // x   bf16 (18432 x 1024)
  unsigned short* wqb = xb + S;                 // Wqkv bf16 (3072 x 1024)
  unsigned short* wpb = wqb + (size_t)3072 * 1024;  // Wproj bf16 (1024 x 1024)

  cvt_bf16<<<18432, 256, 0, stream>>>(x,     xb,  4718592);  // 18432*1024/4
  cvt_bf16<<<3072,  256, 0, stream>>>(Wqkv,  wqb, 786432);
  cvt_bf16<<<1024,  256, 0, stream>>>(Wproj, wpb, 262144);

  qkv_gemm <<<dim3(24, 144), 256, 0, stream>>>(xb, wqb, bqkv, qw, kw, vtw);
  attn_kernel<<<dim3(18, 1024), 32, 0, stream>>>(qw, kw, vtw, yw);
  proj_gemm<<<dim3(8, 144), 256, 0, stream>>>(yw, wpb, bproj, out);
}
